// LSTMEncoder_91070486544428
// MI455X (gfx1250) — compile-verified, matrix operand order checked by reading, not run on hardware
//
#include <hip/hip_runtime.h>
#include <hip/hip_bf16.h>
#include <math.h>

// ---------------------------------------------------------------------------
// Problem constants (reference: B=32, T=512, D=768, H=768, G=4H=3072)
// ---------------------------------------------------------------------------
#define BZ 32
#define TZ 512
#define DZ 768
#define HZ 768
#define GZ 3072
#define MZ (BZ * TZ)          // 16384 rows for the input-projection GEMM

typedef __attribute__((ext_vector_type(16))) __bf16       bf16x16;
typedef __attribute__((ext_vector_type(8)))  float        f32x8;
typedef __attribute__((ext_vector_type(4)))  unsigned int u32x4;
typedef __attribute__((ext_vector_type(8)))  int          i32x8;
typedef __attribute__((ext_vector_type(4)))  int          i32x4;

union Frag16 { bf16x16 v; struct { u32x4 lo, hi; } q; };
union Acc8   { f32x8 v; float f[8]; };

// Load one 16x32 bf16 WMMA operand fragment.  Per ISA 7.12.2:
//   lane L: row = base + (L&15);  half = L>>4
//   elems 0..7  <- k0 + half*8 ;  elems 8..15 <- k0 + 16 + half*8
// => two contiguous 16-byte DS loads per lane.
__device__ inline bf16x16 ld_frag(const __bf16* __restrict base, int ld,
                                  int row_base, int k0) {
  const unsigned lane = threadIdx.x & 31u;
  const int r    = row_base + (int)(lane & 15u);
  const int half = (int)(lane >> 4);
  const __bf16* p = base + (size_t)r * ld + k0 + half * 8;
  Frag16 f;
  f.q.lo = *reinterpret_cast<const u32x4*>(p);
  f.q.hi = *reinterpret_cast<const u32x4*>(p + 16);
  return f.v;
}

__device__ inline f32x8 wmma_bf16(bf16x16 a, bf16x16 b, f32x8 c) {
  return __builtin_amdgcn_wmma_f32_16x16x32_bf16(
      /*neg_a=*/false, a, /*neg_b=*/false, b,
      /*c_mod=*/(short)0, c, /*reuse_a=*/false, /*reuse_b=*/false);
}

__device__ inline float sigmoidf_(float x) { return 1.0f / (1.0f + expf(-x)); }

// ---------------------------------------------------------------------------
// Tensor Data Mover: DMA `rows` x 768 bf16 (rows contiguous in global) into
// LDS at byte offset `lds_off`, with the TDM pad engine inserting 4 dwords of
// padding after every 128 dwords => padded LDS row stride = 396 dwords
// (= RSTRIDE bf16).  396 % 64 = 12 banks, so the 16-lane WMMA fragment
// fan-out (banks 12*m mod 64, all distinct, disjoint 4-bank windows) is
// conflict-free.  D# layout per CDNA5 ISA §8.3/§8.4.
// ---------------------------------------------------------------------------
#define RSTRIDE 792   // padded LDS row stride in bf16 elements (396 dwords)

__device__ inline void tdm_load_rows(unsigned lds_off, const void* gptr,
                                     unsigned rows) {
  const unsigned long long ga = (unsigned long long)(uintptr_t)gptr;
  u32x4 g0;
  g0[0] = 1u;                                        // count=1, user D#
  g0[1] = lds_off;                                   // lds_addr (bytes)
  g0[2] = (unsigned)(ga & 0xffffffffu);              // global_addr[31:0]
  g0[3] = (unsigned)((ga >> 32) & 0x01ffffffu)       // global_addr[56:32]
          | (2u << 30);                              // type = 2 ("image")
  i32x8 g1;
  g1[0] = (int)((1u << 16)        // data_size = 2 bytes
              | (1u << 20)        // pad_enable
              | (6u << 22)        // pad_interval = 128 dwords
              | (3u << 25));      // pad_amount   = 4 dwords
  g1[1] = (int)((768u & 0xffffu) << 16);             // tensor_dim0[15:0]
  g1[2] = (int)((rows & 0xffffu) << 16);             // dim0 hi=0 | tensor_dim1 lo
  g1[3] = (int)(768u << 16);                         // dim1 hi=0 | tile_dim0
  g1[4] = (int)rows;                                 // tile_dim1 (tile_dim2=0)
  g1[5] = (int)768;                                  // tensor_dim0_stride[31:0]
  g1[6] = 0;                                         // stride hi | dim1_stride lo
  g1[7] = 0;
  i32x4 z4 = {0, 0, 0, 0};
#if defined(__clang_major__) && (__clang_major__ >= 23)
  i32x8 z8 = {0, 0, 0, 0, 0, 0, 0, 0};
  __builtin_amdgcn_tensor_load_to_lds(g0, g1, z4, z4, z8, 0);
#else
  __builtin_amdgcn_tensor_load_to_lds(g0, g1, z4, z4, 0);
#endif
}

// Fragment loader for TDM-padded LDS rows: intra-row pads (8 bf16 after every
// 256 bf16) never split a 32-wide K fragment since 32 | 256.
__device__ inline bf16x16 ld_frag_p(const __bf16* __restrict base,
                                    int row_base, int k0) {
  const unsigned lane = threadIdx.x & 31u;
  const int r    = row_base + (int)(lane & 15u);
  const int half = (int)(lane >> 4);
  const int kadj = k0 + ((k0 >> 8) << 3);
  const __bf16* p = base + (size_t)r * RSTRIDE + kadj + half * 8;
  Frag16 f;
  f.q.lo = *reinterpret_cast<const u32x4*>(p);
  f.q.hi = *reinterpret_cast<const u32x4*>(p + 16);
  return f.v;
}

// ---------------------------------------------------------------------------
// Utility kernels
// ---------------------------------------------------------------------------
__global__ __launch_bounds__(256) void k_f32_to_bf16(const float* __restrict in,
                                                     __bf16* __restrict out,
                                                     int n) {
  int i = blockIdx.x * 256 + threadIdx.x;
  if (i < n) out[i] = (__bf16)in[i];
}

// W [K=768][N=3072] f32  ->  Wt [3072][768] bf16
__global__ __launch_bounds__(256) void k_transpose_bf16(const float* __restrict W,
                                                        __bf16* __restrict Wt) {
  int idx = blockIdx.x * 256 + threadIdx.x;   // over K*N
  if (idx >= DZ * GZ) return;
  int k = idx / GZ, n = idx % GZ;
  Wt[(size_t)n * DZ + k] = (__bf16)W[idx];
}

__global__ __launch_bounds__(256) void k_zero_u32(unsigned* __restrict p, int n) {
  int i = blockIdx.x * 256 + threadIdx.x;
  if (i < n) p[i] = 0u;
}

// ---------------------------------------------------------------------------
// Input projection GEMM:  A[M,768] bf16 (rows r = b*T+t) x Bt[3072,768] bf16
//                      -> xg bf16 [T, B, G].  Block tile 64x128, K-step 32.
// ---------------------------------------------------------------------------
__global__ __launch_bounds__(256) void k_gemm_xg(const __bf16* __restrict A,
                                                 const __bf16* __restrict Bt,
                                                 __bf16* __restrict xg) {
  __shared__ __bf16 As[64][40];    // pad 32->40: conflict-free 16-lane fan-out
  __shared__ __bf16 Bs[128][40];

  const int bn  = blockIdx.x * 128;
  const int bm  = blockIdx.y * 64;
  const int tid = threadIdx.x;
  const int w   = tid >> 5;
  const int mi  = (w >> 2) * 32;
  const int ni  = (w & 3) * 32;

  f32x8 acc[2][2] = {};
  const int arow = tid >> 2, acol = (tid & 3) * 8;

  for (int k0 = 0; k0 < DZ; k0 += 32) {
    *reinterpret_cast<u32x4*>(&As[arow][acol]) =
        *reinterpret_cast<const u32x4*>(A + (size_t)(bm + arow) * DZ + k0 + acol);
    #pragma unroll
    for (int i = 0; i < 2; ++i) {
      int idx = tid + i * 256;
      int row = idx >> 2, col = (idx & 3) * 8;
      *reinterpret_cast<u32x4*>(&Bs[row][col]) =
          *reinterpret_cast<const u32x4*>(Bt + (size_t)(bn + row) * DZ + k0 + col);
    }
    if (k0 + 32 < DZ) {   // global_prefetch_b8 on next K-slab
      __builtin_prefetch(A + (size_t)(bm + arow) * DZ + k0 + 32 + acol, 0, 1);
      __builtin_prefetch(Bt + (size_t)(bn + arow) * DZ + k0 + 32 + acol, 0, 1);
    }
    __syncthreads();

    bf16x16 a0 = ld_frag(&As[0][0], 40, mi + 0, 0);
    bf16x16 a1 = ld_frag(&As[0][0], 40, mi + 16, 0);
    bf16x16 b0 = ld_frag(&Bs[0][0], 40, ni + 0, 0);
    bf16x16 b1 = ld_frag(&Bs[0][0], 40, ni + 16, 0);
    acc[0][0] = wmma_bf16(a0, b0, acc[0][0]);
    acc[0][1] = wmma_bf16(a0, b1, acc[0][1]);
    acc[1][0] = wmma_bf16(a1, b0, acc[1][0]);
    acc[1][1] = wmma_bf16(a1, b1, acc[1][1]);
    __syncthreads();
  }

  const unsigned lane = tid & 31u;
  const int half = (int)(lane >> 4), n = (int)(lane & 15u);
  #pragma unroll
  for (int i = 0; i < 2; ++i)
    #pragma unroll
    for (int j = 0; j < 2; ++j) {
      Acc8 ac; ac.v = acc[i][j];
      #pragma unroll
      for (int r = 0; r < 8; ++r) {
        int row = bm + mi + i * 16 + r + half * 8;
        int col = bn + ni + j * 16 + n;
        int b = row >> 9, t = row & (TZ - 1);
        xg[((size_t)t * BZ + b) * GZ + col] = (__bf16)ac.f[r];
      }
    }
}

// ---------------------------------------------------------------------------
// Persistent LSTM recurrence.  48 WGs x 16 h-columns (=> 64 gate columns,
// i/f/g/o slices).  Wh slice and per-step h are DMA'd into LDS by the Tensor
// Data Mover (TENSORcnt); h is double-buffered in global; one grid barrier
// per step.
// ---------------------------------------------------------------------------
#define RWG 48
#define HC  16

__global__ __launch_bounds__(256) void k_lstm_rec(
    const __bf16* __restrict xg,     // [T,B,G] bf16 input projections
    const __bf16* __restrict Wht,    // [G,H]  bf16 (row = gate col)
    const float*  __restrict bias,   // [G] f32
    __bf16* __restrict hbuf,         // [2,B,H] bf16 double-buffered (zeroed)
    __bf16* __restrict hsbf,         // [B,T,H] bf16 layer output
    float*  __restrict hsf,          // [B,T,H] f32  layer output
    unsigned* __restrict barcnt,     // cumulative grid-barrier counter (zeroed)
    int writeF32) {
  extern __shared__ char smem[];
  __bf16* Ws = (__bf16*)smem;                                    // [64][RSTRIDE]
  __bf16* Hs = (__bf16*)(smem + (size_t)64 * RSTRIDE * 2);       // [32][RSTRIDE]
  float*  Gx = (float*)(smem + (size_t)96 * RSTRIDE * 2);        // [4][32][16]

  const int tid = threadIdx.x;
  const int wg  = blockIdx.x;
  const int j0  = wg * HC;

  // LDS byte offsets of the dynamic-LDS partitions (aperture low 32 bits).
  const unsigned ws_off = (unsigned)(uintptr_t)(void*)Ws;
  const unsigned hs_off = (unsigned)(uintptr_t)(void*)Hs;

  // --- TDM: park this WG's Wh slice (4 groups of 16 contiguous gate rows).
  if (tid == 0) {
    #pragma unroll
    for (int q = 0; q < 4; ++q)
      tdm_load_rows(ws_off + (unsigned)(q * 16 * RSTRIDE * 2),
                    Wht + (size_t)(q * HZ + j0) * HZ, 16u);
    __builtin_amdgcn_s_wait_tensorcnt(0);
  }
  __syncthreads();

  float c0 = 0.0f, c1 = 0.0f;    // per-thread cell state (e = tid, tid+256)

  const int w  = tid >> 5;
  const int mi = (w & 1) * 16;   // batch-row tile
  const int qi = w >> 1;         // gate type 0..3 (i,f,g,o)
  const unsigned lane = tid & 31u;
  const int half = (int)(lane >> 4), nn = (int)(lane & 15u);

  for (int s = 0; s < TZ; ++s) {
    const __bf16* hprev = hbuf + (size_t)(s & 1) * (BZ * HZ);
    __bf16*       hnext = hbuf + (size_t)((s + 1) & 1) * (BZ * HZ);

    // --- TDM: stage h_s (L2-resident, 48KB) into LDS
    if (tid == 0) {
      tdm_load_rows(hs_off, hprev, 32u);
      __builtin_amdgcn_s_wait_tensorcnt(0);
    }
    __syncthreads();

    // one 16x16 gate tile per wave, K=768 in 24 WMMA steps
    f32x8 acc = {};
    #pragma unroll 4
    for (int kk = 0; kk < HZ; kk += 32) {
      bf16x16 a = ld_frag_p(Hs, mi, kk);
      bf16x16 b = ld_frag_p(Ws, qi * 16, kk);
      acc = wmma_bf16(a, b, acc);
    }

    // exchange gate tiles through LDS:  Gx[q][brow][j]
    {
      Acc8 ac; ac.v = acc;
      #pragma unroll
      for (int r = 0; r < 8; ++r)
        Gx[(qi * 32 + mi + r + half * 8) * 16 + nn] = ac.f[r];
    }
    __syncthreads();

    // elementwise cell update (2 elements/thread)
    const __bf16* xgs = xg + (size_t)s * BZ * GZ;
    #pragma unroll
    for (int p = 0; p < 2; ++p) {
      int e = tid + p * 256;
      int b = e >> 4, j = e & 15;
      int gc0 = 0 * HZ + j0 + j, gc1 = 1 * HZ + j0 + j;
      int gc2 = 2 * HZ + j0 + j, gc3 = 3 * HZ + j0 + j;
      float gi = Gx[(0 * 32 + b) * 16 + j] + (float)xgs[(size_t)b * GZ + gc0] + bias[gc0];
      float gf = Gx[(1 * 32 + b) * 16 + j] + (float)xgs[(size_t)b * GZ + gc1] + bias[gc1];
      float gg = Gx[(2 * 32 + b) * 16 + j] + (float)xgs[(size_t)b * GZ + gc2] + bias[gc2];
      float go = Gx[(3 * 32 + b) * 16 + j] + (float)xgs[(size_t)b * GZ + gc3] + bias[gc3];
      float iv = sigmoidf_(gi), fv = sigmoidf_(gf);
      float gv = tanhf(gg),     ov = sigmoidf_(go);
      float cpr = p ? c1 : c0;
      float cc = fv * cpr + iv * gv;
      if (p) c1 = cc; else c0 = cc;
      float hh = ov * tanhf(cc);
      int hcol = j0 + j;
      hnext[(size_t)b * HZ + hcol] = (__bf16)hh;
      size_t oidx = ((size_t)b * TZ + s) * HZ + hcol;
      hsbf[oidx] = (__bf16)hh;
      if (writeF32) hsf[oidx] = hh;
    }

    // grid barrier: cumulative counter, s_sleep backoff
    __threadfence();
    __syncthreads();
    if (tid == 0) {
      atomicAdd(barcnt, 1u);
      const unsigned want = (unsigned)(s + 1) * (unsigned)gridDim.x;
      volatile unsigned* vc = barcnt;
      while (*vc < want) __builtin_amdgcn_s_sleep(1);
      __threadfence();
    }
    __syncthreads();
  }
}

// ---------------------------------------------------------------------------
// LayerNorm over H=768: one block per (b,t) row, wave32 shfl reductions.
// ---------------------------------------------------------------------------
__global__ __launch_bounds__(256) void k_layernorm(const float* __restrict X,
                                                   const float* __restrict sc,
                                                   const float* __restrict bi,
                                                   float* __restrict out) {
  __shared__ float red[8];
  const int row = blockIdx.x;
  const int tid = threadIdx.x;
  const float* x = X + (size_t)row * HZ;

  float v0 = x[tid], v1 = x[tid + 256], v2 = x[tid + 512];
  float s = v0 + v1 + v2;
  #pragma unroll
  for (int o = 16; o > 0; o >>= 1) s += __shfl_xor(s, o, 32);
  if ((tid & 31) == 0) red[tid >> 5] = s;
  __syncthreads();
  float mean = 0.0f;
  #pragma unroll
  for (int i = 0; i < 8; ++i) mean += red[i];
  mean *= (1.0f / (float)HZ);
  __syncthreads();

  float d0 = v0 - mean, d1 = v1 - mean, d2 = v2 - mean;
  float vs = d0 * d0 + d1 * d1 + d2 * d2;
  #pragma unroll
  for (int o = 16; o > 0; o >>= 1) vs += __shfl_xor(vs, o, 32);
  if ((tid & 31) == 0) red[tid >> 5] = vs;
  __syncthreads();
  float var = 0.0f;
  #pragma unroll
  for (int i = 0; i < 8; ++i) var += red[i];
  var *= (1.0f / (float)HZ);
  float r = rsqrtf(var + 1e-6f);

  float* o_ = out + (size_t)row * HZ;
  o_[tid]       = d0 * r * sc[tid]       + bi[tid];
  o_[tid + 256] = d1 * r * sc[tid + 256] + bi[tid + 256];
  o_[tid + 512] = d2 * r * sc[tid + 512] + bi[tid + 512];
}

// ---------------------------------------------------------------------------
// Host orchestration
// ---------------------------------------------------------------------------
extern "C" void kernel_launch(void* const* d_in, const int* in_sizes, int n_in,
                              void* d_out, int out_size, void* d_ws, size_t ws_size,
                              hipStream_t stream) {
  (void)in_sizes; (void)n_in; (void)out_size; (void)ws_size;
  const float* x    = (const float*)d_in[0];
  const float* Wx0  = (const float*)d_in[1];
  const float* Wh0  = (const float*)d_in[2];
  const float* b0   = (const float*)d_in[3];
  const float* Wx1  = (const float*)d_in[4];
  const float* Wh1  = (const float*)d_in[5];
  const float* b1   = (const float*)d_in[6];
  const float* lnsc = (const float*)d_in[7];
  const float* lnbi = (const float*)d_in[8];
  float* out = (float*)d_out;
  char*  ws  = (char*)d_ws;

  constexpr size_t SZ_XBF  = (size_t)MZ * DZ * 2;
  constexpr size_t SZ_WT   = (size_t)GZ * DZ * 2;
  constexpr size_t SZ_XG   = (size_t)TZ * BZ * GZ * 2;
  constexpr size_t SZ_HSBF = (size_t)MZ * HZ * 2;
  constexpr size_t SZ_HSF  = (size_t)MZ * HZ * 4;
  constexpr size_t SZ_HBUF = 2ull * BZ * HZ * 2;

  size_t off = 0;
  __bf16*  xbf   = (__bf16*)(ws + off); off += SZ_XBF;
  __bf16*  wx0t  = (__bf16*)(ws + off); off += SZ_WT;
  __bf16*  wh0t  = (__bf16*)(ws + off); off += SZ_WT;
  __bf16*  wx1t  = (__bf16*)(ws + off); off += SZ_WT;
  __bf16*  wh1t  = (__bf16*)(ws + off); off += SZ_WT;
  __bf16*  xgbuf = (__bf16*)(ws + off); off += SZ_XG;
  __bf16*  hsbf  = (__bf16*)(ws + off); off += SZ_HSBF;
  float*   hsf   = (float*)(ws + off);  off += SZ_HSF;
  __bf16*  hbuf  = (__bf16*)(ws + off); off += SZ_HBUF;
  unsigned* cnt  = (unsigned*)(ws + off); off += 256;

  const size_t rec_smem =
      (size_t)96 * RSTRIDE * 2 + 4 * 32 * 16 * sizeof(float);  // 160,256 B

  // 1) precision conversion + weight transposition
  k_f32_to_bf16<<<(MZ * DZ + 255) / 256, 256, 0, stream>>>(x, xbf, MZ * DZ);
  k_transpose_bf16<<<(DZ * GZ + 255) / 256, 256, 0, stream>>>(Wx0, wx0t);
  k_transpose_bf16<<<(DZ * GZ + 255) / 256, 256, 0, stream>>>(Wh0, wh0t);
  k_transpose_bf16<<<(DZ * GZ + 255) / 256, 256, 0, stream>>>(Wx1, wx1t);
  k_transpose_bf16<<<(DZ * GZ + 255) / 256, 256, 0, stream>>>(Wh1, wh1t);

  // 2) layer 0
  {
    int nz = (int)((SZ_HBUF + 256) / 4);
    k_zero_u32<<<(nz + 255) / 256, 256, 0, stream>>>((unsigned*)hbuf, nz);
  }
  k_gemm_xg<<<dim3(GZ / 128, MZ / 64), 256, 0, stream>>>(xbf, wx0t, xgbuf);
  k_lstm_rec<<<RWG, 256, rec_smem, stream>>>(xgbuf, wh0t, b0, hbuf, hsbf, hsf, cnt, 0);

  // 3) layer 1
  {
    int nz = (int)((SZ_HBUF + 256) / 4);
    k_zero_u32<<<(nz + 255) / 256, 256, 0, stream>>>((unsigned*)hbuf, nz);
  }
  k_gemm_xg<<<dim3(GZ / 128, MZ / 64), 256, 0, stream>>>(hsbf, wx1t, xgbuf);
  k_lstm_rec<<<RWG, 256, rec_smem, stream>>>(xgbuf, wh1t, b1, hbuf, hsbf, hsf, cnt, 1);

  // 4) LayerNorm -> d_out
  k_layernorm<<<MZ, 256, 0, stream>>>(hsf, lnsc, lnbi, out);
}
